// MultiHeadSelfAttention_56590489092209
// MI455X (gfx1250) — matrix operand order repaired and checked by reading, not compiled, not run on hardware
//
#include <hip/hip_runtime.h>

// ---------------------------------------------------------------------------
// MHA forward for MI455X (gfx1250): all matmuls via v_wmma_f32_16x16x32_bf16.
// B=4, S=2048, E=1024, H=16, Dh=64.  fp32 in/out, bf16 tensor-core math.
// Attention: scores computed transposed (S^T = K x Q^T) so the WMMA C-layout
// of score tiles is directly the A-operand layout for P x V (no LDS, no
// cross-lane transpose). Online softmax with raw v_exp_f32.
// ---------------------------------------------------------------------------

typedef __bf16 bf16;
typedef __attribute__((ext_vector_type(16))) __bf16 v16bf;
typedef __attribute__((ext_vector_type(8)))  __bf16 v8bf;
typedef __attribute__((ext_vector_type(4)))  __bf16 v4bf;
typedef __attribute__((ext_vector_type(8)))  float  v8f;

#define EMB   1024
#define SEQ   2048
#define BATCH 4
#define HEADS 16
#define HDIM  64
#define MTOT  (BATCH * SEQ)   // 8192 rows

static __device__ inline v16bf join8(v8bf lo, v8bf hi) {
  return __builtin_shufflevector(lo, hi, 0,1,2,3,4,5,6,7,8,9,10,11,12,13,14,15);
}

static __device__ inline v8f wmma_bf16(v16bf a, v16bf b, v8f c) {
  // D = A(16x32 bf16) x B(32x16 bf16) + C(16x16 f32)
  return __builtin_amdgcn_wmma_f32_16x16x32_bf16(
      /*neg_a=*/false, a, /*neg_b=*/false, b,
      /*c_mod=*/(short)0, c, /*reuse_a=*/false, /*reuse_b=*/false);
}

// --------------------------- fp32 -> bf16 convert (x4) ----------------------
__global__ void cvt_f32_bf16(const float* __restrict__ in, bf16* __restrict__ out, int n4) {
  int i = blockIdx.x * blockDim.x + threadIdx.x;
  if (i < n4) {
    float4 v = ((const float4*)in)[i];
    v4bf o = { (bf16)v.x, (bf16)v.y, (bf16)v.z, (bf16)v.w };
    ((v4bf*)out)[i] = o;
  }
}

// ----------------------- V transpose: [B,S,E] -> [B,H,Dh,S] -----------------
__global__ void transpose_v(const bf16* __restrict__ v, bf16* __restrict__ vt) {
  int i = blockIdx.x * blockDim.x + threadIdx.x;  // over B*S*E
  int e = i & (EMB - 1);
  int s = (i >> 10) & (SEQ - 1);
  int b = i >> 21;
  int h = e >> 6;
  int d = e & (HDIM - 1);
  vt[(((size_t)(b * HEADS + h) * HDIM + d) * SEQ) + s] = v[i];
}

// --------------------------- WMMA GEMM: Y = X W^T + b -----------------------
// X: [M,K] bf16 row-major.  W: [N,K] bf16 row-major (so W^T is the B operand).
// One wave computes a 64x64 tile; 8 waves / 256-thread block.
template <int OUTF32>
__global__ __launch_bounds__(256)
void gemm_wmma(const bf16* __restrict__ X, const bf16* __restrict__ W,
               const float* __restrict__ bias,
               bf16* __restrict__ outb, float* __restrict__ outf,
               int M, int N, int K) {
  const int lane = threadIdx.x & 31;
  const int wave = threadIdx.x >> 5;
  const int h    = lane >> 4;    // lane half (0/1)
  const int ln   = lane & 15;

  const int ntiles = N >> 6;
  const int tid = blockIdx.x * 8 + wave;
  const int m0  = (tid / ntiles) << 6;
  const int n0  = (tid % ntiles) << 6;

  v8f acc[4][4];
#pragma unroll
  for (int i = 0; i < 4; ++i)
#pragma unroll
    for (int j = 0; j < 4; ++j) acc[i][j] = (v8f){0.f,0.f,0.f,0.f,0.f,0.f,0.f,0.f};

  for (int k0 = 0; k0 < K; k0 += 32) {
    v16bf a[4];
#pragma unroll
    for (int i = 0; i < 4; ++i) {
      // A layout: lane row m=ln, elems 0..7 at k0+8h, elems 8..15 at k0+8h+16
      const bf16* pa = X + (size_t)(m0 + i * 16 + ln) * K + k0 + 8 * h;
      a[i] = join8(*(const v8bf*)pa, *(const v8bf*)(pa + 16));
    }
#pragma unroll
    for (int j = 0; j < 4; ++j) {
      // B layout: lane col n=ln, elems 0..7 at k0+8h, elems 8..15 at k0+8h+16 (row n of W)
      const bf16* pb = W + (size_t)(n0 + j * 16 + ln) * K + k0 + 8 * h;
      v16bf b = join8(*(const v8bf*)pb, *(const v8bf*)(pb + 16));
#pragma unroll
      for (int i = 0; i < 4; ++i) acc[i][j] = wmma_bf16(a[i], b, acc[i][j]);
    }
  }

#pragma unroll
  for (int j = 0; j < 4; ++j) {
    const float bv = bias[n0 + j * 16 + ln];
    const size_t col = (size_t)(n0 + j * 16 + ln);
#pragma unroll
    for (int i = 0; i < 4; ++i) {
      const int mrow = m0 + i * 16 + 8 * h;   // D layout: VGPR r -> row r+8h
#pragma unroll
      for (int r = 0; r < 8; ++r) {
        float v = acc[i][j][r] + bv;
        size_t off = (size_t)(mrow + r) * N + col;
        if (OUTF32) outf[off] = v; else outb[off] = (bf16)v;
      }
    }
  }
}

// --------------------------- flash attention kernel -------------------------
// 64-key blocks: 4 score tiles -> 2 packed P fragments -> 16 WMMAs per iter,
// one online-softmax pass per block. Raw scores tracked; 1/32 scale folded
// into the exp2 argument (max commutes with positive scaling).
__global__ __launch_bounds__(256)
void attn_wmma(const bf16* __restrict__ Q, const bf16* __restrict__ Kmat,
               const bf16* __restrict__ Vt, bf16* __restrict__ Ctx) {
  const int lane = threadIdx.x & 31;
  const int wave = threadIdx.x >> 5;
  const int h    = lane >> 4;
  const int ln   = lane & 15;

  const int bid = blockIdx.x;          // B*H*16 = 1024 blocks
  const int b   = bid >> 8;
  const int hd  = (bid >> 4) & 15;
  const int q0  = (bid & 15) * 128 + wave * 16;

  const size_t rowbase = (size_t)b * SEQ * EMB + (size_t)hd * HDIM;
  const bf16* vtb = Vt + (size_t)(b * HEADS + hd) * HDIM * SEQ;

  // Q rows as B-operand fragments over the two 32-wide d-halves
  v16bf bq[2];
#pragma unroll
  for (int f = 0; f < 2; ++f) {
    const bf16* p = Q + rowbase + (size_t)(q0 + ln) * EMB + f * 32 + 8 * h;
    bq[f] = join8(*(const v8bf*)p, *(const v8bf*)(p + 16));
  }

  v8f acc[4];
#pragma unroll
  for (int t = 0; t < 4; ++t) acc[t] = (v8f){0.f,0.f,0.f,0.f,0.f,0.f,0.f,0.f};
  float m_run = -1e30f, l_run = 0.f;   // m_run tracks RAW (unscaled) score max

  const float SL2E = 0.03125f * 1.44269504f;   // (1/sqrt(EMB)) * log2(e)

  for (int s0 = 0; s0 < SEQ; s0 += 64) {
    // ---- scores: S^T tiles = K rows (A) x Q rows (B) ----
    v8f st[4];
#pragma unroll
    for (int sb = 0; sb < 4; ++sb) {
      const bf16* p = Kmat + rowbase + (size_t)(s0 + sb * 16 + ln) * EMB + 8 * h;
      v16bf ak0 = join8(*(const v8bf*)p,        *(const v8bf*)(p + 16));
      v16bf ak1 = join8(*(const v8bf*)(p + 32), *(const v8bf*)(p + 48));
      v8f s = (v8f){0.f,0.f,0.f,0.f,0.f,0.f,0.f,0.f};
      s = wmma_bf16(ak0, bq[0], s);
      s = wmma_bf16(ak1, bq[1], s);
      st[sb] = s;
    }

    // ---- online softmax over this 64-key block ----
    float mx = -1e30f;
#pragma unroll
    for (int sb = 0; sb < 4; ++sb)
#pragma unroll
      for (int r = 0; r < 8; ++r) mx = fmaxf(mx, st[sb][r]);
    mx = fmaxf(mx, __shfl_xor(mx, 16, 32));   // combine sk lane-halves
    const float m_new = fmaxf(m_run, mx);

    const float mn2 = m_new * SL2E;
    float lsum = 0.f;
#pragma unroll
    for (int sb = 0; sb < 4; ++sb)
#pragma unroll
      for (int r = 0; r < 8; ++r) {
        float e = __builtin_amdgcn_exp2f(fmaf(st[sb][r], SL2E, -mn2));
        st[sb][r] = e;
        lsum += e;
      }
    lsum += __shfl_xor(lsum, 16, 32);
    const float c = __builtin_amdgcn_exp2f((m_run - m_new) * SL2E);
    l_run = l_run * c + lsum;
    m_run = m_new;

    // rescale accumulators: c indexed by q = r+8h (VGPR dim) -> bpermute
#pragma unroll
    for (int r = 0; r < 8; ++r) {
      float cr = __shfl(c, r + 8 * h, 32);
#pragma unroll
      for (int t = 0; t < 4; ++t) acc[t][r] *= cr;
    }

    // ---- pack P^T tiles into A-operand fragments (in-lane) ----
    v16bf pa0, pa1;
#pragma unroll
    for (int e = 0; e < 8; ++e) {
      pa0[e]     = (bf16)st[0][e];
      pa0[e + 8] = (bf16)st[1][e];
      pa1[e]     = (bf16)st[2][e];
      pa1[e + 8] = (bf16)st[3][e];
    }

    // ---- P x V: V^T rows are contiguous B-operand loads ----
#pragma unroll
    for (int t = 0; t < 4; ++t) {
      const bf16* p = vtb + (size_t)(t * 16 + ln) * SEQ + s0 + 8 * h;
      v16bf bv0 = join8(*(const v8bf*)p,        *(const v8bf*)(p + 16));
      v16bf bv1 = join8(*(const v8bf*)(p + 32), *(const v8bf*)(p + 48));
      acc[t] = wmma_bf16(pa0, bv0, acc[t]);
      acc[t] = wmma_bf16(pa1, bv1, acc[t]);
    }
  }

  // normalize and write context (bf16, [B,S,E] layout for the output GEMM)
  const float linv = __builtin_amdgcn_rcpf(l_run);
#pragma unroll
  for (int r = 0; r < 8; ++r) {
    float nr = __shfl(linv, r + 8 * h, 32);
    const size_t row = (size_t)(q0 + r + 8 * h);
#pragma unroll
    for (int t = 0; t < 4; ++t)
      Ctx[rowbase + row * EMB + t * 16 + ln] = (bf16)(acc[t][r] * nr);
  }
}

// ------------------------------- host launch --------------------------------
extern "C" void kernel_launch(void* const* d_in, const int* in_sizes, int n_in,
                              void* d_out, int out_size, void* d_ws, size_t ws_size,
                              hipStream_t stream) {
  (void)in_sizes; (void)n_in; (void)out_size; (void)ws_size;
  const float* x  = (const float*)d_in[0];
  const float* Wq = (const float*)d_in[1];
  const float* bq = (const float*)d_in[2];
  const float* Wk = (const float*)d_in[3];
  const float* bk = (const float*)d_in[4];
  const float* Wv = (const float*)d_in[5];
  const float* bv = (const float*)d_in[6];
  const float* Wo = (const float*)d_in[7];
  const float* bo = (const float*)d_in[8];
  float* out = (float*)d_out;

  const size_t MB = 1ull << 20;
  char* ws = (char*)d_ws;
  bf16* xb   = (bf16*)(ws +  0 * MB);   // 16 MB  [8192 x 1024]
  bf16* wqb  = (bf16*)(ws + 16 * MB);   //  2 MB
  bf16* wkb  = (bf16*)(ws + 18 * MB);
  bf16* wvb  = (bf16*)(ws + 20 * MB);
  bf16* wob  = (bf16*)(ws + 22 * MB);
  bf16* Qb   = (bf16*)(ws + 24 * MB);   // 16 MB
  bf16* Kb   = (bf16*)(ws + 40 * MB);   // 16 MB
  bf16* Vtmp = (bf16*)(ws + 56 * MB);   // 16 MB (reused as Ctx)
  bf16* Vt   = (bf16*)(ws + 72 * MB);   // 16 MB
  bf16* Ctx  = Vtmp;                    // V no longer needed after transpose

  const int NX = MTOT * EMB;            // 8388608
  const int NW = EMB * EMB;             // 1048576

  cvt_f32_bf16<<<(NX / 4) / 256, 256, 0, stream>>>(x,  xb,  NX / 4);
  cvt_f32_bf16<<<(NW / 4) / 256, 256, 0, stream>>>(Wq, wqb, NW / 4);
  cvt_f32_bf16<<<(NW / 4) / 256, 256, 0, stream>>>(Wk, wkb, NW / 4);
  cvt_f32_bf16<<<(NW / 4) / 256, 256, 0, stream>>>(Wv, wvb, NW / 4);
  cvt_f32_bf16<<<(NW / 4) / 256, 256, 0, stream>>>(Wo, wob, NW / 4);

  // 2048 output tiles of 64x64, 8 waves/block -> 256 blocks
  gemm_wmma<0><<<256, 256, 0, stream>>>(xb, wqb, bq, Qb,   nullptr, MTOT, EMB, EMB);
  gemm_wmma<0><<<256, 256, 0, stream>>>(xb, wkb, bk, Kb,   nullptr, MTOT, EMB, EMB);
  gemm_wmma<0><<<256, 256, 0, stream>>>(xb, wvb, bv, Vtmp, nullptr, MTOT, EMB, EMB);

  transpose_v<<<NX / 256, 256, 0, stream>>>(Vtmp, Vt);

  // B*H*(S/128) = 1024 blocks, 8 waves each handling 16 query rows
  attn_wmma<<<1024, 256, 0, stream>>>(Qb, Kb, Vt, Ctx);

  gemm_wmma<1><<<256, 256, 0, stream>>>(Ctx, wob, bo, nullptr, out, MTOT, EMB, EMB);
}
